// StudentSegContrast_77120432767383
// MI455X (gfx1250) — compile-verified, matrix operand order checked by reading, not run on hardware
//
#include <hip/hip_runtime.h>
#include <hip/hip_bf16.h>
#include <math.h>

typedef __bf16 bf16_t;
typedef __attribute__((ext_vector_type(16))) __bf16 v16bf;
typedef __attribute__((ext_vector_type(8)))  __bf16 v8bf;
typedef __attribute__((ext_vector_type(8)))  float  v8f;

#define NCLS 19
#define NPIX 16384   // B * H * W
#define PPB  4096    // pixels per batch image

// ---------------- workspace layout (bytes) ----------------
static constexpr size_t OFF_TFLAT = 0;                                   // f32 [16384][256]
static constexpr size_t OFF_X     = OFF_TFLAT + (size_t)NPIX * 256 * 4;  // f32 [16384][256] (x, reused as s)
static constexpr size_t OFF_SROW  = OFF_X     + (size_t)NPIX * 256 * 4;  // bf16 [16384][128]
static constexpr size_t OFF_XB    = OFF_SROW  + (size_t)NPIX * 128 * 2;  // bf16 [16384][256]
static constexpr size_t OFF_TBF   = OFF_XB    + (size_t)NPIX * 256 * 2;  // bf16 [16384][256]
static constexpr size_t OFF_SBF   = OFF_TBF   + (size_t)NPIX * 256 * 2;  // bf16 [16384][256]
static constexpr size_t OFF_W1B   = OFF_SBF   + (size_t)NPIX * 256 * 2;  // bf16 [256][128]
static constexpr size_t OFF_W2B   = OFF_W1B   + (size_t)256 * 128 * 2;   // bf16 [256][256]
static constexpr size_t OFF_XPIX  = OFF_W2B   + (size_t)256 * 256 * 2;   // bf16 [4864][256]
static constexpr size_t OFF_XREG  = OFF_XPIX  + (size_t)4864 * 256 * 2;  // bf16 [2432][256]
static constexpr size_t OFF_LAB   = OFF_XREG  + (size_t)2432 * 256 * 2;  // i32 [16384]
static constexpr size_t OFF_TOP   = OFF_LAB   + (size_t)NPIX * 4;        // i32 [76][16]
static constexpr size_t OFF_STATS = OFF_TOP   + (size_t)76 * 16 * 4;     // f32 stats region

// stats region (float indices)
static constexpr int ST_COUNTS = 0;                 // [4*19]
static constexpr int ST_MEANS  = ST_COUNTS + 76;    // [4*19*256]
static constexpr int ST_ACC0   = ST_MEANS + 4 * 19 * 256;
static constexpr int ST_ACC1   = ST_ACC0 + 1;
static constexpr int ST_VALID  = ST_ACC1 + 1;       // int
static constexpr int ST_SCALE  = ST_VALID + 1;      // [256]
static constexpr int ST_SHIFT  = ST_SCALE + 256;    // [256]
static constexpr int ST_TOTAL  = ST_SHIFT + 256;

// ---------------- WMMA helpers ----------------
// 16-bit A 16x32 fragment layout (ISA 7.12.2): lane L -> row (or col) L%16,
// half = L/16; elements 0..7 = K[8*half .. 8*half+7], 8..15 = K[16+8*half ..].
// Both halves are 16B-contiguous in a row-major [row][K] buffer -> two b128 loads.
__device__ __forceinline__ v16bf load_frag(const bf16_t* __restrict__ base, int stride, int lane) {
  int i = lane & 15;
  int h = (lane >> 4) << 3;  // 0 or 8
  const bf16_t* p = base + (size_t)i * stride + h;
  v8bf lo = *reinterpret_cast<const v8bf*>(p);
  v8bf hi = *reinterpret_cast<const v8bf*>(p + 16);
  v16bf r;
#pragma unroll
  for (int e = 0; e < 8; ++e) { r[e] = lo[e]; r[e + 8] = hi[e]; }
  return r;
}

__device__ __forceinline__ v8f wmma_bf16(v16bf a, v16bf b, v8f c) {
  return __builtin_amdgcn_wmma_f32_16x16x32_bf16(false, a, false, b, (short)0, c, false, false);
}

__device__ __forceinline__ float bmax16(float v) {
#pragma unroll
  for (int m = 1; m < 16; m <<= 1) v = fmaxf(v, __shfl_xor(v, m, 16));
  return v;
}
__device__ __forceinline__ float bsum16(float v) {
#pragma unroll
  for (int m = 1; m < 16; m <<= 1) v += __shfl_xor(v, m, 16);
  return v;
}

// ---------------- kernels ----------------
__global__ __launch_bounds__(256) void zero_stats_kernel(float* stats) {
  int i = blockIdx.x * 256 + threadIdx.x;
  if (i < ST_TOTAL) stats[i] = 0.0f;
}

__global__ __launch_bounds__(256) void wconv_kernel(const float* __restrict__ w1,
                                                    const float* __restrict__ w2,
                                                    bf16_t* __restrict__ w1b,
                                                    bf16_t* __restrict__ w2b) {
  int i = blockIdx.x * 256 + threadIdx.x;
  if (i < 256 * 128) w1b[i] = (bf16_t)w1[i];
  if (i < 256 * 256) w2b[i] = (bf16_t)w2[i];
}

// l2-normalize teacher feats over channel dim; (B,C,H,W) -> row-major [p][c]
__global__ __launch_bounds__(256) void tnorm_kernel(const float* __restrict__ tf,
                                                    float* __restrict__ t_flat,
                                                    bf16_t* __restrict__ t_bf) {
  int p = blockIdx.x * 256 + threadIdx.x;
  int b = p >> 12, pix = p & 4095;
  const float* base = tf + (size_t)b * 256 * PPB + pix;
  float ss = 0.f;
  for (int c = 0; c < 256; ++c) { float v = base[(size_t)c * PPB]; ss += v * v; }
  float inv = 1.f / fmaxf(sqrtf(ss), 1e-12f);
  for (int c = 0; c < 256; ++c) {
    float v = base[(size_t)c * PPB] * inv;
    t_flat[(size_t)p * 256 + c] = v;
    t_bf[(size_t)p * 256 + c] = (bf16_t)v;
  }
}

// student feats (B,128,H,W) -> bf16 row-major [p][128]
__global__ __launch_bounds__(256) void sprep_kernel(const float* __restrict__ sf,
                                                    bf16_t* __restrict__ srow) {
  int p = blockIdx.x * 256 + threadIdx.x;
  int b = p >> 12, pix = p & 4095;
  const float* base = sf + (size_t)b * 128 * PPB + pix;
  for (int c = 0; c < 128; ++c) srow[(size_t)p * 128 + c] = (bf16_t)base[(size_t)c * PPB];
}

__global__ __launch_bounds__(256) void label_kernel(const int* __restrict__ labels,
                                                    int* __restrict__ lab,
                                                    float* __restrict__ stats) {
  int p = blockIdx.x * 256 + threadIdx.x;
  int b = p >> 12, hw = p & 4095;
  int h = hw >> 6, w = hw & 63;
  int l = labels[(size_t)b * 512 * 512 + (size_t)h * 8 * 512 + w * 8];
  lab[p] = l;
  if (l != 255) atomicAdd(reinterpret_cast<int*>(stats + ST_VALID), 1);
}

// C[16384][256] = A[16384][K] @ B[256][K]^T   (bf16 WMMA, f32 accum)
template <int K>
__global__ __launch_bounds__(32) void gemm_kernel(const bf16_t* __restrict__ A,
                                                  const bf16_t* __restrict__ B,
                                                  float* __restrict__ C) {
  int lane = threadIdx.x;
  int rowBase = blockIdx.x * 16;
  int n_in = lane & 15;
  int rb = rowBase + ((lane >> 4) << 3);
  // hoist A fragments for this 16-row block
  v16bf afrag[K / 32];
#pragma unroll
  for (int kt = 0; kt < K / 32; ++kt)
    afrag[kt] = load_frag(A + (size_t)rowBase * K + kt * 32, K, lane);
  for (int nt = 0; nt < 16; nt += 2) {
    v8f acc0 = {0.f, 0.f, 0.f, 0.f, 0.f, 0.f, 0.f, 0.f};
    v8f acc1 = {0.f, 0.f, 0.f, 0.f, 0.f, 0.f, 0.f, 0.f};
#pragma unroll
    for (int kt = 0; kt < K / 32; ++kt) {
      v16bf b0 = load_frag(B + (size_t)nt * 16 * K + kt * 32, K, lane);
      v16bf b1 = load_frag(B + (size_t)(nt + 1) * 16 * K + kt * 32, K, lane);
      acc0 = wmma_bf16(afrag[kt], b0, acc0);
      acc1 = wmma_bf16(afrag[kt], b1, acc1);
    }
    int n0 = nt * 16 + n_in;
#pragma unroll
    for (int r = 0; r < 8; ++r) C[(size_t)(rb + r) * 256 + n0] = acc0[r];
#pragma unroll
    for (int r = 0; r < 8; ++r) C[(size_t)(rb + r) * 256 + n0 + 16] = acc1[r];
  }
}

__global__ __launch_bounds__(256) void bn_stats_kernel(const float* __restrict__ x,
                                                       const float* __restrict__ gamma,
                                                       const float* __restrict__ beta,
                                                       float* __restrict__ stats) {
  __shared__ float s1[256], s2[256];
  int o = blockIdx.x, t = threadIdx.x;
  float a = 0.f, b = 0.f;
  for (int p = t; p < NPIX; p += 256) {
    float v = x[(size_t)p * 256 + o];
    a += v; b += v * v;
  }
  s1[t] = a; s2[t] = b;
  __syncthreads();
  for (int s = 128; s > 0; s >>= 1) {
    if (t < s) { s1[t] += s1[t + s]; s2[t] += s2[t + s]; }
    __syncthreads();
  }
  if (t == 0) {
    float mu = s1[0] / (float)NPIX;
    float var = s2[0] / (float)NPIX - mu * mu;
    float sc = gamma[o] * rsqrtf(var + 1e-5f);
    stats[ST_SCALE + o] = sc;
    stats[ST_SHIFT + o] = beta[o] - mu * sc;
  }
}

__global__ __launch_bounds__(256) void bnact_kernel(const float* __restrict__ x,
                                                    const float* __restrict__ stats,
                                                    bf16_t* __restrict__ xb) {
  size_t i = (size_t)blockIdx.x * 256 + threadIdx.x;
  int o = (int)(i & 255);
  float v = fmaxf(x[i] * stats[ST_SCALE + o] + stats[ST_SHIFT + o], 0.f);
  xb[i] = (bf16_t)v;
}

__global__ __launch_bounds__(256) void snorm_kernel(const float* __restrict__ s,
                                                    bf16_t* __restrict__ sb) {
  int p = blockIdx.x * 256 + threadIdx.x;
  const float* row = s + (size_t)p * 256;
  float ss = 0.f;
  for (int c = 0; c < 256; ++c) { float v = row[c]; ss += v * v; }
  float inv = 1.f / fmaxf(sqrtf(ss), 1e-12f);
  for (int c = 0; c < 256; ++c) sb[(size_t)p * 256 + c] = (bf16_t)(row[c] * inv);
}

// per (batch,class) sum of t_n rows + counts (atomic accumulation)
__global__ __launch_bounds__(256) void accum_means_kernel(const int* __restrict__ lab,
                                                          const float* __restrict__ t_flat,
                                                          float* __restrict__ stats) {
  int p = blockIdx.x;
  int cls = lab[p];
  if (cls == 255) return;
  int b = p >> 12;
  atomicAdd(&stats[ST_MEANS + ((b * NCLS + cls) * 256) + threadIdx.x],
            t_flat[(size_t)p * 256 + threadIdx.x]);
  if (threadIdx.x == 0) atomicAdd(&stats[ST_COUNTS + b * NCLS + cls], 1.0f);
}

// top-16 pixels by pix_rand within each (b, cls) — iterative argmax, ties -> smallest index
__global__ __launch_bounds__(256) void top16_kernel(const int* __restrict__ lab,
                                                    const float* __restrict__ pix_rand,
                                                    int* __restrict__ top_idx) {
  __shared__ float sc[PPB];
  __shared__ float bv[256];
  __shared__ int bi[256];
  int b = blockIdx.x / NCLS, cls = blockIdx.x % NCLS;
  int t = threadIdx.x;
  for (int i = 0; i < 16; ++i) {
    int p = t + i * 256;
    sc[p] = (lab[b * PPB + p] == cls) ? pix_rand[b * PPB + p] : -1e30f;
  }
  __syncthreads();
  for (int round = 0; round < 16; ++round) {
    float v = -INFINITY; int idx = PPB;
    for (int i = 0; i < 16; ++i) {
      int p = t * 16 + i;
      float x = sc[p];
      if (x > v) { v = x; idx = p; }
    }
    bv[t] = v; bi[t] = idx;
    __syncthreads();
    for (int s = 128; s > 0; s >>= 1) {
      if (t < s) {
        if (bv[t + s] > bv[t] || (bv[t + s] == bv[t] && bi[t + s] < bi[t])) {
          bv[t] = bv[t + s]; bi[t] = bi[t + s];
        }
      }
      __syncthreads();
    }
    if (t == 0) {
      top_idx[blockIdx.x * 16 + round] = bi[0];
      sc[bi[0]] = -INFINITY;
    }
    __syncthreads();
  }
}

// X_reg[cls*128 + j] = (idx<4 && present) ? l2norm(mean[b=idx][cls]) : seg_q[cls][idx]
__global__ __launch_bounds__(256) void seg_build_kernel(const int* __restrict__ region_index,
                                                        const float* __restrict__ seg_q,
                                                        const float* __restrict__ stats,
                                                        bf16_t* __restrict__ xreg) {
  __shared__ float red[256];
  int cls = blockIdx.x >> 7, j = blockIdx.x & 127;
  int d = threadIdx.x;
  int idx = region_index[j];
  float out;
  bool use_new = false;
  float val = 0.f, cnt = 0.f;
  if (idx < 4) {
    cnt = stats[ST_COUNTS + idx * NCLS + cls];
    if (cnt > 0.f) {
      use_new = true;
      val = stats[ST_MEANS + ((idx * NCLS + cls) * 256) + d] / fmaxf(cnt, 1.0f);
    }
  }
  if (use_new) {
    red[d] = val * val;
    __syncthreads();
    for (int s = 128; s > 0; s >>= 1) { if (d < s) red[d] += red[d + s]; __syncthreads(); }
    float inv = 1.f / fmaxf(sqrtf(red[0]), 1e-12f);
    out = val * inv;
  } else {
    out = seg_q[((size_t)cls * 2000 + idx) * 256 + d];
  }
  xreg[(size_t)blockIdx.x * 256 + d] = (bf16_t)out;
}

// X_pix[cls*256 + j] = (idx<64 && present) ? t_n[selected pixel] : pix_q[cls][idx]
__global__ __launch_bounds__(256) void pix_build_kernel(const int* __restrict__ pixel_index,
                                                        const float* __restrict__ pix_q,
                                                        const float* __restrict__ t_flat,
                                                        const int* __restrict__ top_idx,
                                                        const float* __restrict__ stats,
                                                        bf16_t* __restrict__ xpix) {
  int cls = blockIdx.x >> 8, j = blockIdx.x & 255;
  int d = threadIdx.x;
  int idx = pixel_index[j];
  float out;
  bool use_new = false;
  int b = 0, r = 0;
  if (idx < 64) {
    b = idx >> 4; r = idx & 15;
    if (stats[ST_COUNTS + b * NCLS + cls] > 0.f) use_new = true;
  }
  if (use_new) {
    int pix = top_idx[(b * NCLS + cls) * 16 + r];
    out = t_flat[((size_t)(b * PPB + pix)) * 256 + d];  // t_n rows already unit norm
  } else {
    out = pix_q[((size_t)cls * 20000 + idx) * 256 + d];
  }
  xpix[(size_t)blockIdx.x * 256 + d] = (bf16_t)out;
}

// Fused dual-GEMM + streaming log-softmax KL. One wave owns 16 rows; iterates
// column tiles of X two at a time (4 independent WMMA accumulator chains to
// absorb the bf16 WMMA->WMMA hazard window) keeping online (max, Z, W) per row.
__global__ __launch_bounds__(32) void kd_kernel(const bf16_t* __restrict__ At,
                                                const bf16_t* __restrict__ As,
                                                const bf16_t* __restrict__ X,
                                                int ncols,
                                                const int* __restrict__ lab,
                                                float* __restrict__ acc_out) {
  int lane = threadIdx.x;
  int rowBase = blockIdx.x * 16;
  int half = lane >> 4;
  float m_t[8], z_t[8], w_[8], m_s[8], z_s[8];
#pragma unroll
  for (int r = 0; r < 8; ++r) { m_t[r] = -INFINITY; m_s[r] = -INFINITY; z_t[r] = 0.f; z_s[r] = 0.f; w_[r] = 0.f; }

  // hoist both A-matrix fragment sets (t and s): 16 frags = 128 VGPRs, live
  // across the whole column loop.
  v16bf at[8], as[8];
#pragma unroll
  for (int kt = 0; kt < 8; ++kt) {
    at[kt] = load_frag(At + (size_t)rowBase * 256 + kt * 32, 256, lane);
    as[kt] = load_frag(As + (size_t)rowBase * 256 + kt * 32, 256, lane);
  }

  int ntiles = ncols >> 4;  // 304 or 152, both even
  for (int ct = 0; ct < ntiles; ct += 2) {
    v8f dt0 = {0.f, 0.f, 0.f, 0.f, 0.f, 0.f, 0.f, 0.f};
    v8f dt1 = {0.f, 0.f, 0.f, 0.f, 0.f, 0.f, 0.f, 0.f};
    v8f ds0 = {0.f, 0.f, 0.f, 0.f, 0.f, 0.f, 0.f, 0.f};
    v8f ds1 = {0.f, 0.f, 0.f, 0.f, 0.f, 0.f, 0.f, 0.f};
#pragma unroll
    for (int kt = 0; kt < 8; ++kt) {
      v16bf b0 = load_frag(X + (size_t)ct * 16 * 256 + kt * 32, 256, lane);
      v16bf b1 = load_frag(X + (size_t)(ct + 1) * 16 * 256 + kt * 32, 256, lane);
      dt0 = wmma_bf16(at[kt], b0, dt0);
      dt1 = wmma_bf16(at[kt], b1, dt1);
      ds0 = wmma_bf16(as[kt], b0, ds0);
      ds1 = wmma_bf16(as[kt], b1, ds1);
    }
#pragma unroll
    for (int u = 0; u < 2; ++u) {
      v8f dt = u ? dt1 : dt0;
      v8f ds = u ? ds1 : ds0;
#pragma unroll
      for (int r = 0; r < 8; ++r) {
        float tv = dt[r] * 10.0f;  // / T_C
        float sv = ds[r] * 10.0f;
        float nm_t = fmaxf(m_t[r], bmax16(tv));
        float nm_s = fmaxf(m_s[r], bmax16(sv));
        float pt = __expf(tv - nm_t);
        float ps = __expf(sv - nm_s);
        float sp = bsum16(pt);
        float spw = bsum16(pt * (tv - sv));
        float sps = bsum16(ps);
        float corr_t = __expf(m_t[r] - nm_t);
        float corr_s = __expf(m_s[r] - nm_s);
        z_t[r] = z_t[r] * corr_t + sp;
        w_[r]  = w_[r]  * corr_t + spw;
        z_s[r] = z_s[r] * corr_s + sps;
        m_t[r] = nm_t; m_s[r] = nm_s;
      }
    }
  }
  // stats replicated across the 16 lanes of each half; lanes 0 and 16 finalize
  if ((lane & 15) == 0) {
    float tot = 0.f;
#pragma unroll
    for (int r = 0; r < 8; ++r) {
      int row = rowBase + r + 8 * half;
      if (lab[row] != 255) {
        float lse_t = m_t[r] + __logf(z_t[r]);
        float lse_s = m_s[r] + __logf(z_s[r]);
        tot += w_[r] / z_t[r] - lse_t + lse_s;
      }
    }
    atomicAdd(acc_out, tot);
  }
}

__global__ void finalize_kernel(const float* __restrict__ stats, float* __restrict__ out) {
  int nv = reinterpret_cast<const int*>(stats)[ST_VALID];
  float n = (float)(nv < 1 ? 1 : nv);
  out[0] = stats[ST_ACC0] / n;  // pixel_sim_dis (T_KD^2 == 1)
  out[1] = stats[ST_ACC1] / n;  // region_sim_dis
}

// ---------------- launcher ----------------
extern "C" void kernel_launch(void* const* d_in, const int* in_sizes, int n_in,
                              void* d_out, int out_size, void* d_ws, size_t ws_size,
                              hipStream_t stream) {
  (void)in_sizes; (void)n_in; (void)out_size; (void)ws_size;
  const float* s_feats = (const float*)d_in[0];
  const float* t_feats = (const float*)d_in[1];
  const float* w1      = (const float*)d_in[2];
  const float* gamma   = (const float*)d_in[3];
  const float* beta    = (const float*)d_in[4];
  const float* w2      = (const float*)d_in[5];
  const float* seg_q   = (const float*)d_in[6];
  const float* pix_q   = (const float*)d_in[7];
  const int*   labels  = (const int*)d_in[8];
  const int*   pix_idx = (const int*)d_in[9];
  const int*   reg_idx = (const int*)d_in[10];
  const float* pixrand = (const float*)d_in[11];
  float* out = (float*)d_out;

  char* ws = (char*)d_ws;
  float*  t_flat = (float*)(ws + OFF_TFLAT);
  float*  xbuf   = (float*)(ws + OFF_X);
  bf16_t* srow   = (bf16_t*)(ws + OFF_SROW);
  bf16_t* xb     = (bf16_t*)(ws + OFF_XB);
  bf16_t* t_bf   = (bf16_t*)(ws + OFF_TBF);
  bf16_t* s_bf   = (bf16_t*)(ws + OFF_SBF);
  bf16_t* w1b    = (bf16_t*)(ws + OFF_W1B);
  bf16_t* w2b    = (bf16_t*)(ws + OFF_W2B);
  bf16_t* xpix   = (bf16_t*)(ws + OFF_XPIX);
  bf16_t* xreg   = (bf16_t*)(ws + OFF_XREG);
  int*    lab    = (int*)(ws + OFF_LAB);
  int*    topidx = (int*)(ws + OFF_TOP);
  float*  stats  = (float*)(ws + OFF_STATS);

  zero_stats_kernel<<<(ST_TOTAL + 255) / 256, 256, 0, stream>>>(stats);
  wconv_kernel<<<256, 256, 0, stream>>>(w1, w2, w1b, w2b);
  tnorm_kernel<<<NPIX / 256, 256, 0, stream>>>(t_feats, t_flat, t_bf);
  sprep_kernel<<<NPIX / 256, 256, 0, stream>>>(s_feats, srow);
  label_kernel<<<NPIX / 256, 256, 0, stream>>>(labels, lab, stats);

  // student head: GEMM1 -> BN stats -> BN+ReLU -> GEMM2 -> row l2norm
  gemm_kernel<128><<<NPIX / 16, 32, 0, stream>>>(srow, w1b, xbuf);
  bn_stats_kernel<<<256, 256, 0, stream>>>(xbuf, gamma, beta, stats);
  bnact_kernel<<<NPIX, 256, 0, stream>>>(xbuf, stats, xb);
  gemm_kernel<256><<<NPIX / 16, 32, 0, stream>>>(xb, w2b, xbuf);  // xbuf now holds s
  snorm_kernel<<<NPIX / 256, 256, 0, stream>>>(xbuf, s_bf);

  // queue updates (only affect gathered X rows)
  accum_means_kernel<<<NPIX, 256, 0, stream>>>(lab, t_flat, stats);
  top16_kernel<<<4 * NCLS, 256, 0, stream>>>(lab, pixrand, topidx);
  seg_build_kernel<<<NCLS * 128, 256, 0, stream>>>(reg_idx, seg_q, stats, xreg);
  pix_build_kernel<<<NCLS * 256, 256, 0, stream>>>(pix_idx, pix_q, t_flat, topidx, stats, xpix);

  // fused dual-GEMM + streaming KL (flash-style; never materializes logits)
  kd_kernel<<<NPIX / 16, 32, 0, stream>>>(t_bf, s_bf, xpix, NCLS * 256, lab, stats + ST_ACC0);
  kd_kernel<<<NPIX / 16, 32, 0, stream>>>(t_bf, s_bf, xreg, NCLS * 128, lab, stats + ST_ACC1);

  finalize_kernel<<<1, 1, 0, stream>>>(stats, out);
}
